// VariationalGCNConvEncoder_55379308315092
// MI455X (gfx1250) — compile-verified, matrix operand order checked by reading, not run on hardware
//
#include <hip/hip_runtime.h>

// ---------------------------------------------------------------------------
// Types for CDNA5 WMMA (wave32): v_wmma_f32_16x16x32_bf16
// ---------------------------------------------------------------------------
typedef __bf16 bf16_t;
typedef bf16_t v16bf __attribute__((ext_vector_type(16)));
typedef float  v8f   __attribute__((ext_vector_type(8)));

union FragU {
    v16bf v;
    uint4 q[2];
};

__device__ __forceinline__ bf16_t f2bf(float f) {
    // round-to-nearest-even f32 -> bf16
    unsigned u = __float_as_uint(f);
    unsigned r = (u + 0x7FFFu + ((u >> 16) & 1u)) >> 16;
    unsigned short s = (unsigned short)r;
    bf16_t b;
    __builtin_memcpy(&b, &s, 2);
    return b;
}

// ---------------------------------------------------------------------------
// Degree / normalization kernels
// ---------------------------------------------------------------------------
__global__ void k_deg_init(float* __restrict__ deg, int n) {
    int i = blockIdx.x * blockDim.x + threadIdx.x;
    if (i < n) deg[i] = 1.0f;   // self-loop weight 1
}

__global__ void k_deg_edges(const long long* __restrict__ col,
                            const float* __restrict__ ew,
                            float* __restrict__ deg, int E) {
    int e = blockIdx.x * blockDim.x + threadIdx.x;
    if (e < E) atomicAdd(&deg[(int)col[e]], ew[e]);
}

__global__ void k_rsqrt_inplace(float* __restrict__ d, int n) {
    int i = blockIdx.x * blockDim.x + threadIdx.x;
    if (i < n) {
        float v = d[i];
        d[i] = (v > 0.0f) ? rsqrtf(v) : 0.0f;
    }
}

// ---------------------------------------------------------------------------
// Aggregation: dst = Anorm @ src  (self-loop init + edge scatter-add)
// ---------------------------------------------------------------------------
__global__ void k_self_init(const float* __restrict__ src,
                            const float* __restrict__ dinv,
                            float* __restrict__ dst, int Nn, int D) {
    int q = D >> 2;
    int idx = blockIdx.x * blockDim.x + threadIdx.x;
    if (idx >= Nn * q) return;
    int row = idx / q;
    float s = dinv[row];
    s = s * s;                          // norm of self-loop: dinv*1*dinv
    float4 v = ((const float4*)src)[idx];
    v.x *= s; v.y *= s; v.z *= s; v.w *= s;
    ((float4*)dst)[idx] = v;
}

// one wave (32 lanes) per edge; each lane handles D/32 elems via float4
__global__ void k_edge_agg(const long long* __restrict__ rowIdx,
                           const long long* __restrict__ colIdx,
                           const float* __restrict__ ew,
                           const float* __restrict__ dinv,
                           const float* __restrict__ src,
                           float* __restrict__ dst,
                           int E, int D) {
    int lane = threadIdx.x & 31;
    int e = (blockIdx.x * blockDim.x + threadIdx.x) >> 5;
    if (e >= E) return;
    int r = (int)rowIdx[e];
    int c = (int)colIdx[e];
    float w = ew[e] * dinv[r] * dinv[c];
    const float4* s4 = (const float4*)(src + (size_t)r * D);
    float* d = dst + (size_t)c * D;
    int q = D >> 2;
    for (int j = lane; j < q; j += 32) {
        float4 v = s4[j];
        atomicAdd(&d[j * 4 + 0], v.x * w);
        atomicAdd(&d[j * 4 + 1], v.y * w);
        atomicAdd(&d[j * 4 + 2], v.z * w);
        atomicAdd(&d[j * 4 + 3], v.w * w);
    }
}

// ---------------------------------------------------------------------------
// Conversions
// ---------------------------------------------------------------------------
__global__ void k_cvt_bf16(const float* __restrict__ in,
                           bf16_t* __restrict__ out, int n) {
    int i = blockIdx.x * blockDim.x + threadIdx.x;
    if (i < n) out[i] = f2bf(in[i]);
}

// W [K,N] row-major f32 -> Wt [N,K] row-major bf16
__global__ void k_wt_bf16(const float* __restrict__ W,
                          bf16_t* __restrict__ Wt, int K, int N) {
    int idx = blockIdx.x * blockDim.x + threadIdx.x;
    if (idx >= K * N) return;
    int n = idx % N;
    int k = idx / N;
    Wt[(size_t)n * K + k] = f2bf(W[idx]);
}

// ---------------------------------------------------------------------------
// WMMA GEMM: Out[M,N] = A[M,K](bf16) * Bt[N,K]^T(bf16) + bias, optional ReLU
// One wave computes a 16x64 tile (A-fragment reused across 4 accumulators).
// ---------------------------------------------------------------------------
__device__ __forceinline__ void wmma_epilogue(float* __restrict__ Out,
                                              v8f c, float bn,
                                              int tm, int half, int ncol,
                                              int Nld, int doRelu) {
#pragma unroll
    for (int r = 0; r < 8; ++r) {
        float v = c[r] + bn;
        if (doRelu) v = fmaxf(v, 0.0f);
        Out[(size_t)(tm + r + half * 8) * Nld + ncol] = v;
    }
}

__global__ void k_gemm_bf16_wmma(const bf16_t* __restrict__ A,
                                 const bf16_t* __restrict__ Bt,
                                 const float* __restrict__ bias,
                                 float* __restrict__ Out,
                                 int M, int N, int K, int doRelu) {
    const int lane = threadIdx.x & 31;
    const int wv = threadIdx.x >> 5;
    const int tile = blockIdx.x * (blockDim.x >> 5) + wv;
    const int gN = N >> 6;                      // 64-col groups
    const int total = (M >> 4) * gN;
    if (tile >= total) return;                  // wave-uniform: EXEC stays all-1s
    const int tm = (tile / gN) << 4;
    const int tn = (tile % gN) << 6;

    const int half = lane >> 4;                 // 0: k 0-7,16-23; 1: k 8-15,24-31
    const int l16 = lane & 15;

    const bf16_t* arow  = A  + (size_t)(tm + l16) * K;
    const bf16_t* brow0 = Bt + (size_t)(tn +  0 + l16) * K;
    const bf16_t* brow1 = Bt + (size_t)(tn + 16 + l16) * K;
    const bf16_t* brow2 = Bt + (size_t)(tn + 32 + l16) * K;
    const bf16_t* brow3 = Bt + (size_t)(tn + 48 + l16) * K;

    v8f c0 = {}; v8f c1 = {}; v8f c2 = {}; v8f c3 = {};

    for (int kt = 0; kt < K; kt += 32) {
        const int kb = kt + half * 8;           // lane's K base within 32-chunk
        FragU a, b0, b1, b2, b3;
        a.q[0]  = *(const uint4*)(arow  + kb);
        a.q[1]  = *(const uint4*)(arow  + kb + 16);
        b0.q[0] = *(const uint4*)(brow0 + kb);
        b0.q[1] = *(const uint4*)(brow0 + kb + 16);
        b1.q[0] = *(const uint4*)(brow1 + kb);
        b1.q[1] = *(const uint4*)(brow1 + kb + 16);
        b2.q[0] = *(const uint4*)(brow2 + kb);
        b2.q[1] = *(const uint4*)(brow2 + kb + 16);
        b3.q[0] = *(const uint4*)(brow3 + kb);
        b3.q[1] = *(const uint4*)(brow3 + kb + 16);

        c0 = __builtin_amdgcn_wmma_f32_16x16x32_bf16(false, a.v, false, b0.v,
                                                     (short)0, c0, false, false);
        c1 = __builtin_amdgcn_wmma_f32_16x16x32_bf16(false, a.v, false, b1.v,
                                                     (short)0, c1, false, false);
        c2 = __builtin_amdgcn_wmma_f32_16x16x32_bf16(false, a.v, false, b2.v,
                                                     (short)0, c2, false, false);
        c3 = __builtin_amdgcn_wmma_f32_16x16x32_bf16(false, a.v, false, b3.v,
                                                     (short)0, c3, false, false);
    }

    wmma_epilogue(Out, c0, bias[tn +  0 + l16], tm, half, tn +  0 + l16, N, doRelu);
    wmma_epilogue(Out, c1, bias[tn + 16 + l16], tm, half, tn + 16 + l16, N, doRelu);
    wmma_epilogue(Out, c2, bias[tn + 32 + l16], tm, half, tn + 32 + l16, N, doRelu);
    wmma_epilogue(Out, c3, bias[tn + 48 + l16], tm, half, tn + 48 + l16, N, doRelu);
}

// ---------------------------------------------------------------------------
// Host launcher
// ---------------------------------------------------------------------------
static inline size_t align256(size_t v) { return (v + 255) & ~(size_t)255; }

extern "C" void kernel_launch(void* const* d_in, const int* in_sizes, int n_in,
                              void* d_out, int out_size, void* d_ws, size_t ws_size,
                              hipStream_t stream) {
    const float*     x    = (const float*)d_in[0];
    const long long* eidx = (const long long*)d_in[1];   // [2, E] int64
    const float*     ew   = (const float*)d_in[2];
    const float*     W1   = (const float*)d_in[3];
    const float*     b1   = (const float*)d_in[4];
    const float*     Wmu  = (const float*)d_in[5];
    const float*     bmu  = (const float*)d_in[6];
    const float*     Wls  = (const float*)d_in[7];
    const float*     bls  = (const float*)d_in[8];
    float* out = (float*)d_out;

    const int E    = in_sizes[2];
    const int DH   = in_sizes[4];                 // 256
    const int DOUT = in_sizes[6];                 // 128
    const int DIN  = in_sizes[3] / DH;            // 256
    const int Nn   = in_sizes[0] / DIN;           // 50000
    const int Dmax = (DIN > DH) ? DIN : DH;

    const long long* rowI = eidx;
    const long long* colI = eidx + E;

    // ---- workspace carve-out ----
    char* w = (char*)d_ws;
    size_t off = 0;
    float*  dinv = (float*)(w + off);  off += align256((size_t)Nn * 4);
    float*  bufA = (float*)(w + off);  off += align256((size_t)Nn * Dmax * 4);   // aggX then aggH (f32)
    bf16_t* bufB = (bf16_t*)(w + off); off += align256((size_t)Nn * Dmax * 2);   // bf16 copy of agg
    float*  hbuf = (float*)(w + off);  off += align256((size_t)Nn * DH * 4);     // relu(GCN1) activations
    bf16_t* W1t  = (bf16_t*)(w + off); off += align256((size_t)DIN * DH * 2);
    bf16_t* Wmut = (bf16_t*)(w + off); off += align256((size_t)DH * DOUT * 2);
    bf16_t* Wlst = (bf16_t*)(w + off); off += align256((size_t)DH * DOUT * 2);
    (void)ws_size; (void)n_in; (void)out_size;

    const int T = 256;
    auto cdiv = [](long long a, long long b) { return (int)((a + b - 1) / b); };

    // ---- weight prep (independent of graph work) ----
    k_wt_bf16<<<cdiv((long long)DIN * DH, T), T, 0, stream>>>(W1, W1t, DIN, DH);
    k_wt_bf16<<<cdiv((long long)DH * DOUT, T), T, 0, stream>>>(Wmu, Wmut, DH, DOUT);
    k_wt_bf16<<<cdiv((long long)DH * DOUT, T), T, 0, stream>>>(Wls, Wlst, DH, DOUT);

    // ---- degree + symmetric normalization ----
    k_deg_init<<<cdiv(Nn, T), T, 0, stream>>>(dinv, Nn);
    k_deg_edges<<<cdiv(E, T), T, 0, stream>>>(colI, ew, dinv, E);
    k_rsqrt_inplace<<<cdiv(Nn, T), T, 0, stream>>>(dinv, Nn);

    // ---- layer 1: aggX = Anorm @ x ; h = relu(aggX @ W1 + b1) ----
    k_self_init<<<cdiv((long long)Nn * (DIN >> 2), T), T, 0, stream>>>(x, dinv, bufA, Nn, DIN);
    k_edge_agg<<<cdiv((long long)E * 32, T), T, 0, stream>>>(rowI, colI, ew, dinv, x, bufA, E, DIN);
    k_cvt_bf16<<<cdiv((long long)Nn * DIN, T), T, 0, stream>>>(bufA, bufB, Nn * DIN);
    {
        int waves = (Nn >> 4) * (DH >> 6);
        k_gemm_bf16_wmma<<<cdiv(waves, T >> 5), T, 0, stream>>>(bufB, W1t, b1, hbuf,
                                                                Nn, DH, DIN, 1);
    }

    // ---- layers 2+3 share one aggregation: aggH = Anorm @ h ----
    k_self_init<<<cdiv((long long)Nn * (DH >> 2), T), T, 0, stream>>>(hbuf, dinv, bufA, Nn, DH);
    k_edge_agg<<<cdiv((long long)E * 32, T), T, 0, stream>>>(rowI, colI, ew, dinv, hbuf, bufA, E, DH);
    k_cvt_bf16<<<cdiv((long long)Nn * DH, T), T, 0, stream>>>(bufA, bufB, Nn * DH);
    {
        int waves = (Nn >> 4) * (DOUT >> 6);
        // mu -> d_out[0 : N*DOUT)
        k_gemm_bf16_wmma<<<cdiv(waves, T >> 5), T, 0, stream>>>(bufB, Wmut, bmu, out,
                                                                Nn, DOUT, DH, 0);
        // logstd -> d_out[N*DOUT : 2*N*DOUT)
        k_gemm_bf16_wmma<<<cdiv(waves, T >> 5), T, 0, stream>>>(bufB, Wlst, bls,
                                                                out + (size_t)Nn * DOUT,
                                                                Nn, DOUT, DH, 0);
    }
}
